// OmostDenseDiffusionCrossAttention_22703197127020
// MI455X (gfx1250) — matrix operand order repaired and checked by reading, not compiled, not run on hardware
//
#include <hip/hip_runtime.h>
#include <math.h>

// ---------------------------------------------------------------------------
// Masked cross-attention (Omost DenseDiffusion), MI455X / gfx1250.
// HBM-bound on mask_scale (165MB f32) + mask_bool (41MB u8): ~280MB total
// -> ~12us floor at 23.3 TB/s. Math (10.3 GFLOP) via v_wmma_f32_16x16x32_f16.
//   - K fragments read straight from global (L2-resident) -> no DScnt false
//     dependency against score staging stores, no K preload.
//   - 163072B LDS per 4-wave WG -> exactly 2 WGs / WGP -> 2 waves per SIMD.
//   - mask stream: row-contiguous float4/uint32, branch-free, prefetched.
//   - only one workgroup barrier (Vt preload); wave-private phases use
//     wave_barrier only.
// ---------------------------------------------------------------------------

typedef __attribute__((ext_vector_type(16))) _Float16 v16h;
typedef __attribute__((ext_vector_type(4)))  _Float16 v4h;
typedef __attribute__((ext_vector_type(8)))  float    v8f;

namespace {
constexpr int BB    = 4;
constexpr int SQ    = 4096;
constexpr int SK    = 308;
constexpr int HEADS = 8;
constexpr int DH    = 64;
constexpr int INNER = HEADS * DH;

constexpr int SKP   = 320;            // keys padded to 20 tiles of 16
constexpr int NKT   = SKP / 16;       // 20 key tiles
constexpr int WAVES = 4;
constexpr int TPB   = WAVES * 32;     // 128 threads
constexpr int QTILE = 16;             // queries per wave
constexpr int WGQ   = WAVES * QTILE;  // 64 queries per workgroup
constexpr int QBLKS = SQ / WGQ;       // 64 q-blocks per (b,h)
constexpr int KSPLIT = 152;           // column split point (mult of 4)

// LDS layout (bytes) -- total 163072 = 80 x 2KB blocks = half a WGP
constexpr int LDS_VT  = 0;                         // f16 [DH][SKP] (V transposed)
constexpr int SROWP   = 316;                       // f32 per score row (>=308,
                                                   //  16B-aligned, 8*316%64=32)
constexpr int SBYTES  = QTILE * SROWP * 4;         // 20224 per wave
constexpr int LDS_S   = LDS_VT + DH * SKP * 2;     // 40960
constexpr int PBYTES  = QTILE * SKP * 2;           // 10240 per wave
constexpr int LDS_P   = LDS_S + WAVES * SBYTES;    // 121856
constexpr int LDS_R   = LDS_P + WAVES * PBYTES;    // 162816
constexpr int LDS_TOTAL = LDS_R + WAVES * QTILE * 4;  // 163072 bytes
} // namespace

struct U128x2 { uint4 lo; uint4 hi; };
union FragCast { U128x2 u; v16h v; };

__device__ __forceinline__ v16h cvt16(const float4 f0, const float4 f1,
                                      const float4 f2, const float4 f3) {
    v16h a;
    a[0]  = (_Float16)f0.x; a[1]  = (_Float16)f0.y;
    a[2]  = (_Float16)f0.z; a[3]  = (_Float16)f0.w;
    a[4]  = (_Float16)f1.x; a[5]  = (_Float16)f1.y;
    a[6]  = (_Float16)f1.z; a[7]  = (_Float16)f1.w;
    a[8]  = (_Float16)f2.x; a[9]  = (_Float16)f2.y;
    a[10] = (_Float16)f2.z; a[11] = (_Float16)f2.w;
    a[12] = (_Float16)f3.x; a[13] = (_Float16)f3.y;
    a[14] = (_Float16)f3.z; a[15] = (_Float16)f3.w;
    return a;
}

__global__ __launch_bounds__(TPB)
void omost_dd_xattn_kernel(const float* __restrict__ qg,
                           const float* __restrict__ kg,
                           const float* __restrict__ vg,
                           const float* __restrict__ msg,
                           const unsigned char* __restrict__ mbg,
                           float* __restrict__ outg) {
    extern __shared__ char smem[];

    const int bh = blockIdx.x / QBLKS;       // 0..31
    const int tq = blockIdx.x % QBLKS;       // 0..63
    const int b  = bh / HEADS;
    const int h  = bh % HEADS;

    const int wave = threadIdx.x >> 5;
    const int lane = threadIdx.x & 31;
    const bool hi  = lane >= 16;
    const int  ln  = lane & 15;

    const float NEG_INF = -__builtin_inff();
    const int q0 = tq * WGQ + wave * QTILE;

    // -----------------------------------------------------------------
    // Prefetch this wave's mask rows (the dominant HBM stream) so they
    // travel while we preload V and run the QK WMMAs. -> global_prefetch_b8
    // -----------------------------------------------------------------
    {
        const size_t rowbase = ((size_t)bh * SQ + q0) * (size_t)SK;
        const char* mp = (const char*)(msg + rowbase);
        const char* bp = (const char*)(mbg + rowbase);
        for (int off = lane * 128; off < QTILE * SK * 4; off += 32 * 128)
            __builtin_prefetch(mp + off, 0, 1);
        for (int off = lane * 128; off < QTILE * SK; off += 32 * 128)
            __builtin_prefetch(bp + off, 0, 1);
    }

    // -----------------------------------------------------------------
    // Preload V -> LDS f16 transposed [DH][SKP]. Pad keys clamp to key
    // SK-1 (finite garbage is fine: P is exactly 0 on pad columns).
    // -----------------------------------------------------------------
    _Float16* Vtl = (_Float16*)(smem + LDS_VT);
    for (int idx = threadIdx.x; idx < SKP * (DH / 4); idx += TPB) {
        const int key  = idx / (DH / 4);
        const int d4   = (idx % (DH / 4)) * 4;
        const int keyc = key < SK ? key : SK - 1;
        const float4 fv = *(const float4*)(vg + ((size_t)b * SK + keyc) * INNER
                                              + h * DH + d4);
        Vtl[(d4 + 0) * SKP + key] = (_Float16)fv.x;
        Vtl[(d4 + 1) * SKP + key] = (_Float16)fv.y;
        Vtl[(d4 + 2) * SKP + key] = (_Float16)fv.z;
        Vtl[(d4 + 3) * SKP + key] = (_Float16)fv.w;
    }
    __syncthreads();   // only workgroup-wide barrier in the kernel

    // -----------------------------------------------------------------
    // Q A-fragments (16x32 f16): lane M = ln; halves j<8 -> K=32c+o+j,
    // j>=8 -> K=32c+o+16+(j-8), o = hi ? 8 : 0.
    // -----------------------------------------------------------------
    v16h aq[2];
    {
        const float* qrow = qg + ((size_t)b * SQ + q0 + ln) * INNER + h * DH;
        const int o = hi ? 8 : 0;
#pragma unroll
        for (int c = 0; c < 2; ++c) {
            const int base = 32 * c + o;
            aq[c] = cvt16(*(const float4*)(qrow + base),
                          *(const float4*)(qrow + base + 4),
                          *(const float4*)(qrow + base + 16),
                          *(const float4*)(qrow + base + 20));
        }
    }

    // -----------------------------------------------------------------
    // QK^T: B-fragment of K^T comes straight from global K (L2-resident):
    // lane owns key 16t+ln, reads two 8-dim runs (o..o+7, o+16..o+23) per chunk.
    // Global loads (LOADcnt) don't serialize against score ds_stores
    // (DScnt). Raw scores staged in LDS; pad columns never stored.
    // -----------------------------------------------------------------
    float* Srow = (float*)(smem + LDS_S + wave * SBYTES);
    for (int t = 0; t < NKT; ++t) {
        const int kk   = t * 16 + ln;
        const int keyc = kk < SK ? kk : SK - 1;           // clamp pad keys
        const float* krow = kg + ((size_t)b * SK + keyc) * INNER + h * DH;
        v8f acc = {};
#pragma unroll
        for (int c = 0; c < 2; ++c) {
            const int base = 32 * c + (hi ? 8 : 0);       // dims o..o+7, o+16..o+23
            const v16h bk = cvt16(*(const float4*)(krow + base),
                                  *(const float4*)(krow + base + 4),
                                  *(const float4*)(krow + base + 16),
                                  *(const float4*)(krow + base + 20));
            acc = __builtin_amdgcn_wmma_f32_16x16x32_f16(
                false, aq[c], false, bk, (short)0, acc, false, false);
        }
        if (kk < SK) {                                    // exec-masked stores
#pragma unroll
            for (int i = 0; i < 8; ++i) {
                const int m = i + (hi ? 8 : 0);
                Srow[m * SROWP + kk] = acc[i];
            }
        }
    }
    __builtin_amdgcn_wave_barrier();   // S is wave-private; LDS in-order/wave

    // -----------------------------------------------------------------
    // Mask + softmax, all 32 lanes: lane ln owns query row ln; lane halves
    // split columns [0,152) / [152,308). mask_scale rows: 1232B (16B
    // aligned, 77 exact float4s); mask_bool rows: 308B (77 uint32s).
    // -----------------------------------------------------------------
    {
        const int kbeg = hi ? KSPLIT : 0;
        const int kend = hi ? SK : KSPLIT;
        const size_t rowoff = ((size_t)bh * SQ + q0 + ln) * (size_t)SK;
        const float*         msrow = msg + rowoff;
        const unsigned char* mbrow = mbg + rowoff;
        float* Sr = Srow + ln * SROWP;

        float mx = NEG_INF;
#pragma unroll 4
        for (int k2 = kbeg; k2 < kend; k2 += 4) {
            const float4 s4 = *(const float4*)(Sr + k2);
            const float4 m4 = *(const float4*)(msrow + k2);
            const unsigned int b4 = *(const unsigned int*)(mbrow + k2);
            float4 r;
            r.x = (b4 & 0x000000ffu) ? s4.x * 0.125f * m4.x : NEG_INF;
            r.y = (b4 & 0x0000ff00u) ? s4.y * 0.125f * m4.y : NEG_INF;
            r.z = (b4 & 0x00ff0000u) ? s4.z * 0.125f * m4.z : NEG_INF;
            r.w = (b4 & 0xff000000u) ? s4.w * 0.125f * m4.w : NEG_INF;
            *(float4*)(Sr + k2) = r;
            mx = fmaxf(mx, fmaxf(fmaxf(r.x, r.y), fmaxf(r.z, r.w)));
        }
        mx = fmaxf(mx, __shfl_xor(mx, 16, 32));

        _Float16* Pr = (_Float16*)(smem + LDS_P + wave * PBYTES) + ln * SKP;
        float sum = 0.f;
#pragma unroll 4
        for (int k2 = kbeg; k2 < kend; k2 += 4) {
            const float4 f = *(const float4*)(Sr + k2);
            const float e0 = __expf(f.x - mx);
            const float e1 = __expf(f.y - mx);
            const float e2 = __expf(f.z - mx);
            const float e3 = __expf(f.w - mx);
            sum += (e0 + e1) + (e2 + e3);
            v4h p; p[0] = (_Float16)e0; p[1] = (_Float16)e1;
                   p[2] = (_Float16)e2; p[3] = (_Float16)e3;
            *(v4h*)(Pr + k2) = p;
        }
        if (hi) {  // zero padded P columns 308..319 so PV pad tiles are clean
            const v4h z = {};
            *(v4h*)(Pr + 308) = z;
            *(v4h*)(Pr + 312) = z;
            *(v4h*)(Pr + 316) = z;
        }
        sum += __shfl_xor(sum, 16, 32);
        if (!hi) ((float*)(smem + LDS_R))[wave * QTILE + ln] = 1.0f / sum;
    }
    __builtin_amdgcn_wave_barrier();   // P/rsum are wave-private

    // -----------------------------------------------------------------
    // Out = P~ x V : 10 key-chunks (K=32) x 4 dim-tiles (N=16).
    // A-frag of P: two 16B LDS loads; B-frag of V^T: one 32B LDS load.
    // -----------------------------------------------------------------
    const char* Pb  = smem + LDS_P + wave * PBYTES;
    const char* Vb  = smem + LDS_VT;
    v8f oacc[4] = {v8f{}, v8f{}, v8f{}, v8f{}};
    for (int kc = 0; kc < SKP / 32; ++kc) {
        FragCast fa;
        const char* ap = Pb + ln * (SKP * 2) + 64 * kc + (hi ? 32 : 0);
        fa.u.lo = *(const uint4*)ap;
        fa.u.hi = *(const uint4*)(ap + 16);
        const v16h aP = fa.v;
#pragma unroll
        for (int nt = 0; nt < 4; ++nt) {
            const v16h bv = *(const v16h*)(Vb + ((size_t)(nt * 16 + ln)) * (SKP * 2)
                                              + 64 * kc + (hi ? 32 : 0));
            oacc[nt] = __builtin_amdgcn_wmma_f32_16x16x32_f16(
                false, aP, false, bv, (short)0, oacc[nt], false, false);
        }
    }

    const float* Rs = (const float*)(smem + LDS_R) + wave * QTILE;
#pragma unroll
    for (int nt = 0; nt < 4; ++nt) {
#pragma unroll
        for (int i = 0; i < 8; ++i) {
            const int m = i + (hi ? 8 : 0);
            const int d = nt * 16 + ln;
            outg[((size_t)b * SQ + q0 + m) * INNER + h * DH + d] = oacc[nt][i] * Rs[m];
        }
    }
}

extern "C" void kernel_launch(void* const* d_in, const int* in_sizes, int n_in,
                              void* d_out, int out_size, void* d_ws, size_t ws_size,
                              hipStream_t stream) {
    (void)in_sizes; (void)n_in; (void)out_size; (void)d_ws; (void)ws_size;
    const float* q  = (const float*)d_in[0];
    const float* k  = (const float*)d_in[1];
    const float* v  = (const float*)d_in[2];
    const float* ms = (const float*)d_in[3];
    const unsigned char* mb = (const unsigned char*)d_in[4];  // numpy bool: 1 byte
    float* out = (float*)d_out;

    hipFuncSetAttribute((const void*)omost_dd_xattn_kernel,
                        hipFuncAttributeMaxDynamicSharedMemorySize, LDS_TOTAL);

    dim3 grid(BB * HEADS * QBLKS);   // 2048 workgroups
    dim3 block(TPB);                 // 128 threads = 4 wave32
    omost_dd_xattn_kernel<<<grid, block, LDS_TOTAL, stream>>>(q, k, v, ms, mb, out);
}
